// GATHAConv_25778393711289
// MI455X (gfx1250) — matrix operand order repaired and checked by reading, not compiled, not run on hardware
//
#include <hip/hip_runtime.h>
#include <hip/hip_bf16.h>
#include <math.h>

#define N_NODES   100000
#define N_EDGES   1600000
#define IN_F      128
#define HEADS     2
#define OUT_F     64
#define HF        (HEADS * OUT_F)   /* 128 */
#define K_HOPS    3
#define NEG_SLOPE 0.2f

typedef __attribute__((ext_vector_type(16))) _Float16 v16h;
typedef __attribute__((ext_vector_type(8)))  float    v8f;

__device__ __forceinline__ float leaky(float x) {
    return x >= 0.0f ? x : NEG_SLOPE * x;
}

__device__ __forceinline__ void atomicMaxF(float* addr, float val) {
    unsigned int* ua = (unsigned int*)addr;
    unsigned int old = __hip_atomic_load(ua, __ATOMIC_RELAXED, __HIP_MEMORY_SCOPE_AGENT);
    while (__uint_as_float(old) < val) {
        unsigned int assumed = old;
        old = atomicCAS(ua, assumed, __float_as_uint(val));
        if (old == assumed) break;
    }
}

// ---------------------------------------------------------------- init / utils
__global__ void k_init(float* msrc, float* mdst, float* ssrc, float* sdst,
                       float* odeg, float* ideg) {
    int i = blockIdx.x * blockDim.x + threadIdx.x;
    if (i < N_NODES * HEADS) {
        msrc[i] = -3.0e38f; mdst[i] = -3.0e38f;
        ssrc[i] = 0.0f;     sdst[i] = 0.0f;
    }
    if (i < N_NODES) { odeg[i] = 0.0f; ideg[i] = 0.0f; }
}

__global__ void k_convw(const float* __restrict__ W, _Float16* __restrict__ Wh) {
    int i = blockIdx.x * blockDim.x + threadIdx.x;
    if (i < HF * IN_F) Wh[i] = (_Float16)W[i];
}

__global__ void k_zero(float* p, int n) {
    int i = blockIdx.x * blockDim.x + threadIdx.x;
    if (i < n) p[i] = 0.0f;
}

// ---------------------------------------------------------------- WMMA GEMM
// h[n][j] = sum_k feat[n][k] * W[j][k]   (A = feat, B[k][j] = W[j][k])
// Block: 256 threads = 8 waves; wave w computes 16x16 tile at
// rows [blockIdx.x*16, +16), cols [w*16, +16). K = 128 -> 4 WMMA f16 steps.
__global__ __launch_bounds__(256)
void k_gemm_fc(const float* __restrict__ feat, const _Float16* __restrict__ Wh,
               float* __restrict__ h) {
    __shared__ _Float16 sW[HF * IN_F];   // 32 KB  [j][k]
    __shared__ _Float16 sA[16 * IN_F];   //  4 KB  [m][k]
    const int tid   = threadIdx.x;
    const int mtile = blockIdx.x;        // 0..6249, 6250*16 == 100000 exactly

    for (int i = tid; i < HF * IN_F; i += 256) sW[i] = Wh[i];
    for (int i = tid; i < 16 * IN_F; i += 256) {
        int m = i >> 7, k = i & 127;
        sA[i] = (_Float16)feat[(size_t)(mtile * 16 + m) * IN_F + k];
    }
    __syncthreads();

    const int wave  = tid >> 5;          // = ntile (0..7)
    const int lane  = tid & 31;
    const int ml    = lane & 15;
    const int khalf = (lane >> 4) * 16;  // A-matrix lane-half K offset
    const int ncol  = wave * 16 + ml;    // B/C column carried by this lane

    v8f c = {};
#pragma unroll
    for (int kk = 0; kk < 4; ++kk) {
        const int k0 = kk * 32;
        v16h a, b;
#pragma unroll
        for (int j = 0; j < 8; ++j) {
            a[j]     = sA[ml * IN_F + k0 + khalf + j];          // K = k0+khalf+j
            a[j + 8] = sA[ml * IN_F + k0 + khalf + 8 + j];      // K = k0+khalf+8+j
        }
        const int kb = k0 + (lane >> 4) * 16;                   // B: K base per lane-half
#pragma unroll
        for (int j = 0; j < 16; ++j)
            b[j] = sW[ncol * IN_F + kb + j];                    // B[k][n] = W[n][k]
        c = __builtin_amdgcn_wmma_f32_16x16x32_f16(
                false, a, false, b, (short)0, c, false, false);
    }

    const int mbase = mtile * 16 + (lane >> 4) * 8;             // C: M = v + half*8
#pragma unroll
    for (int v = 0; v < 8; ++v)
        h[(size_t)(mbase + v) * HF + ncol] = c[v];
}

// ---------------------------------------------------------------- node dots
__global__ void k_node_dots(const float* __restrict__ h,
                            const float* __restrict__ attn_l,
                            const float* __restrict__ attn_r,
                            const float* __restrict__ hop_l,
                            float* el, float* er, float* al) {
    int i = blockIdx.x * blockDim.x + threadIdx.x;
    if (i >= N_NODES * HEADS) return;
    int n = i >> 1, hh = i & 1;
    const float* hp = h + (size_t)n * HF + hh * OUT_F;
    const float* wl = attn_l + hh * OUT_F;
    const float* wr = attn_r + hh * OUT_F;
    const float* wa = hop_l  + hh * OUT_F;
    float sl = 0.f, sr = 0.f, sa = 0.f;
    for (int f = 0; f < OUT_F; ++f) {
        float v = hp[f];
        sl += v * wl[f]; sr += v * wr[f]; sa += v * wa[f];
    }
    el[i] = sl; er[i] = sr; al[i] = sa;
}

// ---------------------------------------------------------------- edge passes
__global__ void k_edge_max(const int* __restrict__ src, const int* __restrict__ dst,
                           const float* __restrict__ el, const float* __restrict__ er,
                           float* msrc, float* mdst, float* odeg, float* ideg) {
    int i = blockIdx.x * blockDim.x + threadIdx.x;
    if (i >= N_EDGES) return;
    int s = src[i], d = dst[i];
#pragma unroll
    for (int hh = 0; hh < HEADS; ++hh) {
        float ev = leaky(el[s * HEADS + hh] + er[d * HEADS + hh]);
        atomicMaxF(&msrc[s * HEADS + hh], ev);
        atomicMaxF(&mdst[d * HEADS + hh], ev);
    }
    atomicAdd(&odeg[s], 1.0f);
    atomicAdd(&ideg[d], 1.0f);
}

__global__ void k_edge_expsum(const int* __restrict__ src, const int* __restrict__ dst,
                              const float* __restrict__ el, const float* __restrict__ er,
                              const float* __restrict__ msrc, const float* __restrict__ mdst,
                              float* ssrc, float* sdst) {
    int i = blockIdx.x * blockDim.x + threadIdx.x;
    if (i >= N_EDGES) return;
    int s = src[i], d = dst[i];
#pragma unroll
    for (int hh = 0; hh < HEADS; ++hh) {
        float ev = leaky(el[s * HEADS + hh] + er[d * HEADS + hh]);
        atomicAdd(&ssrc[s * HEADS + hh], expf(ev - msrc[s * HEADS + hh]));
        atomicAdd(&sdst[d * HEADS + hh], expf(ev - mdst[d * HEADS + hh]));
    }
}

__global__ void k_norms(const float* odeg, const float* ideg,
                        float* nout, float* nin) {
    int i = blockIdx.x * blockDim.x + threadIdx.x;
    if (i >= N_NODES) return;
    nout[i] = rsqrtf(fmaxf(odeg[i], 1.0f));   // out_deg^-0.5
    nin[i]  = sqrtf(fmaxf(ideg[i], 1.0f));    // in_deg^0.5
}

__global__ void k_edge_alpha(const int* __restrict__ src, const int* __restrict__ dst,
                             const float* __restrict__ el, const float* __restrict__ er,
                             const float* __restrict__ msrc, const float* __restrict__ mdst,
                             const float* __restrict__ ssrc, const float* __restrict__ sdst,
                             const float* __restrict__ nout, const float* __restrict__ sigma,
                             float* __restrict__ wedge) {
    int i = blockIdx.x * blockDim.x + threadIdx.x;
    if (i >= N_EDGES) return;
    float sg = 1.0f / (1.0f + expf(-sigma[0]));
    int s = src[i], d = dst[i];
    float no = nout[s];
#pragma unroll
    for (int hh = 0; hh < HEADS; ++hh) {
        float ev = leaky(el[s * HEADS + hh] + er[d * HEADS + hh]);
        float as = fmaxf(expf(ev - msrc[s * HEADS + hh]) / ssrc[s * HEADS + hh], 1e-10f);
        float ad = fmaxf(expf(ev - mdst[d * HEADS + hh]) / sdst[d * HEADS + hh], 1e-10f);
        wedge[i * HEADS + hh] = powf(ad, sg) * powf(as, 1.0f - sg) * no;
    }
}

// ---------------------------------------------------------------- propagation
// thread per (edge, feature-of-128): ft_next[dst] += ft[src] * (a * norm_out[src])
__global__ __launch_bounds__(256)
void k_scatter(const int* __restrict__ src, const int* __restrict__ dst,
               const float* __restrict__ wedge,
               const float* __restrict__ ftp, float* __restrict__ ftn) {
    int t = blockIdx.x * blockDim.x + threadIdx.x;   // < N_EDGES * 128
    int e  = t >> 7;
    int hf = t & 127;
    int hh = hf >> 6;
    int s = src[e], d = dst[e];
    float v = ftp[(size_t)s * HF + hf] * wedge[e * HEADS + hh];
    atomicAdd(&ftn[(size_t)d * HF + hf], v);
}

__global__ void k_scale_in(float* ft, const float* __restrict__ nin) {
    int i = blockIdx.x * blockDim.x + threadIdx.x;
    if (i >= N_NODES * HF) return;
    ft[i] *= nin[i >> 7];
}

// ---------------------------------------------------------------- hop fusion
__global__ void k_fuse(const float* __restrict__ h0, const float* __restrict__ h1,
                       const float* __restrict__ h2, const float* __restrict__ h3,
                       const float* __restrict__ al, const float* __restrict__ hop_r,
                       float* __restrict__ out) {
    int i = blockIdx.x * blockDim.x + threadIdx.x;
    if (i >= N_NODES * HEADS) return;
    int n = i >> 1, hh = i & 1;
    size_t base = (size_t)n * HF + hh * OUT_F;
    const float* hs[K_HOPS + 1] = { h0 + base, h1 + base, h2 + base, h3 + base };
    const float* wr = hop_r + hh * OUT_F;
    float a0 = al[i];

    float lg[K_HOPS + 1];
    float mx = -3.0e38f;
#pragma unroll
    for (int k = 0; k <= K_HOPS; ++k) {
        float d = 0.f;
        for (int f = 0; f < OUT_F; ++f) d += hs[k][f] * wr[f];
        lg[k] = leaky(a0 + d);
        mx = fmaxf(mx, lg[k]);
    }
    float s = 0.f;
#pragma unroll
    for (int k = 0; k <= K_HOPS; ++k) { lg[k] = expf(lg[k] - mx); s += lg[k]; }
    float inv = 1.0f / s;

    float* op = out + base;
    for (int f = 0; f < OUT_F; ++f) {
        float acc = 0.f;
#pragma unroll
        for (int k = 0; k <= K_HOPS; ++k) acc += hs[k][f] * lg[k];
        op[f] = acc * inv;
    }
}

// ---------------------------------------------------------------- launcher
extern "C" void kernel_launch(void* const* d_in, const int* in_sizes, int n_in,
                              void* d_out, int out_size, void* d_ws, size_t ws_size,
                              hipStream_t stream) {
    const float* feat   = (const float*)d_in[0];
    const int*   src    = (const int*)  d_in[1];
    const int*   dst    = (const int*)  d_in[2];
    const float* W_fc   = (const float*)d_in[3];
    const float* attn_l = (const float*)d_in[4];
    const float* attn_r = (const float*)d_in[5];
    const float* hop_l  = (const float*)d_in[6];
    const float* hop_r  = (const float*)d_in[7];
    const float* sigma  = (const float*)d_in[8];

    char* ws = (char*)d_ws;
    size_t off = 0;
    auto carve = [&](size_t bytes) -> void* {
        void* p = (void*)(ws + off);
        off += (bytes + 255) & ~(size_t)255;
        return p;
    };

    _Float16* Wh  = (_Float16*)carve((size_t)HF * IN_F * sizeof(_Float16));
    float* hbuf[K_HOPS + 1];
    for (int k = 0; k <= K_HOPS; ++k)
        hbuf[k] = (float*)carve((size_t)N_NODES * HF * sizeof(float));
    float* el   = (float*)carve((size_t)N_NODES * HEADS * sizeof(float));
    float* er   = (float*)carve((size_t)N_NODES * HEADS * sizeof(float));
    float* al   = (float*)carve((size_t)N_NODES * HEADS * sizeof(float));
    float* msrc = (float*)carve((size_t)N_NODES * HEADS * sizeof(float));
    float* mdst = (float*)carve((size_t)N_NODES * HEADS * sizeof(float));
    float* ssrc = (float*)carve((size_t)N_NODES * HEADS * sizeof(float));
    float* sdst = (float*)carve((size_t)N_NODES * HEADS * sizeof(float));
    float* odeg = (float*)carve((size_t)N_NODES * sizeof(float));
    float* ideg = (float*)carve((size_t)N_NODES * sizeof(float));
    float* nout = (float*)carve((size_t)N_NODES * sizeof(float));
    float* nin  = (float*)carve((size_t)N_NODES * sizeof(float));
    float* wedge= (float*)carve((size_t)N_EDGES * HEADS * sizeof(float));

    const int T = 256;
    const int gNH   = (N_NODES * HEADS + T - 1) / T;
    const int gN    = (N_NODES + T - 1) / T;
    const int gE    = (N_EDGES + T - 1) / T;
    const int gNF   = (N_NODES * HF + T - 1) / T;
    const int gEF   = (N_EDGES * (HF / 2) * HEADS) / T;   // E*128/256 = 800000
    const int gGemm = N_NODES / 16;                       // 6250 (exact)
    const int gW    = (HF * IN_F + T - 1) / T;

    k_init<<<gNH, T, 0, stream>>>(msrc, mdst, ssrc, sdst, odeg, ideg);
    k_convw<<<gW, T, 0, stream>>>(W_fc, Wh);
    k_gemm_fc<<<gGemm, T, 0, stream>>>(feat, Wh, hbuf[0]);
    k_node_dots<<<gNH, T, 0, stream>>>(hbuf[0], attn_l, attn_r, hop_l, el, er, al);
    k_edge_max<<<gE, T, 0, stream>>>(src, dst, el, er, msrc, mdst, odeg, ideg);
    k_edge_expsum<<<gE, T, 0, stream>>>(src, dst, el, er, msrc, mdst, ssrc, sdst);
    k_norms<<<gN, T, 0, stream>>>(odeg, ideg, nout, nin);
    k_edge_alpha<<<gE, T, 0, stream>>>(src, dst, el, er, msrc, mdst, ssrc, sdst,
                                       nout, sigma, wedge);
    for (int k = 1; k <= K_HOPS; ++k) {
        k_zero<<<gNF, T, 0, stream>>>(hbuf[k], N_NODES * HF);
        k_scatter<<<gEF, T, 0, stream>>>(src, dst, wedge, hbuf[k - 1], hbuf[k]);
        k_scale_in<<<gNF, T, 0, stream>>>(hbuf[k], nin);
    }
    k_fuse<<<gNH, T, 0, stream>>>(hbuf[0], hbuf[1], hbuf[2], hbuf[3],
                                  al, hop_r, (float*)d_out);
}
